// MP_PDE_Solver2D_88347477279247
// MI455X (gfx1250) — matrix operand order repaired and checked by reading, not compiled, not run on hardware
//
#include <hip/hip_runtime.h>
#include <cstddef>

#define N_NODES 32768
#define N_GRAPHS 8
#define NODES_PER_GRAPH 4096
#define DEG 8
#define N_EDGES (N_NODES * DEG)
#define TW 25
#define HDIM 128
#define N_LAYERS 6
#define L_PDE 16.0f
#define TMAX_C 4.0f
#define DT_C 0.04f
#define EPS_C 1e-5f

typedef __attribute__((ext_vector_type(16))) _Float16 v16h;
typedef __attribute__((ext_vector_type(8)))  float    v8f;
typedef __attribute__((ext_vector_type(4)))  int      v4i_t;

__device__ __forceinline__ float swishf(float x) { return x / (1.0f + __expf(-x)); }

// ---------------------------------------------------------------------------
// CDNA5 async global->LDS staging (GLOBAL_LOAD_ASYNC_TO_LDS_B128, ASYNCcnt).
// Builtin takes (global int4*, local int4*, imm offset, imm cpol).
// Falls back to a synchronous VGPR round-trip if the builtin is unavailable.
// ---------------------------------------------------------------------------
#if defined(__HIP_DEVICE_COMPILE__) && __has_builtin(__builtin_amdgcn_global_load_async_to_lds_b128)
#define HAVE_ASYNC_LDS 1
#else
#define HAVE_ASYNC_LDS 0
#endif

typedef __attribute__((address_space(1))) v4i_t* gv4p_t;
typedef __attribute__((address_space(3))) v4i_t* lv4p_t;

// stage 8 halves (16 bytes) from global to LDS
__device__ __forceinline__ void stage8(const _Float16* g, _Float16* l) {
#if HAVE_ASYNC_LDS
  __builtin_amdgcn_global_load_async_to_lds_b128((gv4p_t)g, (lv4p_t)l, 0, 0);
#else
  *(float4*)l = *(const float4*)g;
#endif
}

__device__ __forceinline__ void wait_async() {
#if HAVE_ASYNC_LDS
#if __has_builtin(__builtin_amdgcn_s_wait_asynccnt)
  __builtin_amdgcn_s_wait_asynccnt(0);
#else
  asm volatile("s_wait_asynccnt 0x0" ::: "memory");
#endif
#endif
}

// ---------------------------------------------------------------------------
// WMMA fragment loaders (CDNA5 16x16x32 f16, wave32).
// A (16x32 MxK): lanes 0-15 -> M=lane, K=0..15;
//                lanes 16-31 -> M=lane-16, K=16..31.
// B (32x16 KxN) staged transposed [N][K]: lanes 0-15 -> N=lane, K=0..15;
//                lanes 16-31 -> N=lane-16, K=16..31.
// C/D (16x16 f32): vgpr r -> M = r + (lane<16 ? 0 : 8), N = lane & 15.
// ---------------------------------------------------------------------------
__device__ __forceinline__ v16h load_a_frag(const _Float16* sA, int lda, int lane) {
  int row = lane & 15;
  int klo = (lane < 16) ? 0 : 16;
  int khi = (lane < 16) ? 8 : 24;
  const _Float16* p = sA + row * lda;
  v16h a;
#pragma unroll
  for (int i = 0; i < 8; ++i) a[i] = p[klo + i];
#pragma unroll
  for (int i = 0; i < 8; ++i) a[8 + i] = p[khi + i];
  return a;
}

__device__ __forceinline__ v16h load_b_frag(const _Float16* sBt, int ldb, int lane) {
  int n = lane & 15;
  int k0 = (lane < 16) ? 0 : 16;
  const _Float16* p = sBt + n * ldb + k0;
  v16h b;
#pragma unroll
  for (int i = 0; i < 16; ++i) b[i] = p[i];
  return b;
}

// ---------------------------------------------------------------------------
// Generic WMMA GEMM: C[M,Ncols] = swish(A[M,K](f16) @ B[K,Ncols](f16) + bias)
// B supplied transposed as Bt[Ncols][K]. Block = 256 threads = 8 waves,
// computes 128 rows x 128 cols (grid.y selects 128-col slab).
// ---------------------------------------------------------------------------
template <bool HASF, bool HASH>
__global__ __launch_bounds__(256)
void k_gemm(const _Float16* __restrict__ A, int lda,
            const _Float16* __restrict__ Bt, int ldb,
            const float* __restrict__ bias,
            float* __restrict__ Cf, int ldcf,
            _Float16* __restrict__ Ch, int ldch,
            int K) {
  __shared__ __align__(16) _Float16 sA[128 * 40];
  __shared__ __align__(16) _Float16 sB[128 * 40];
  const int tid = threadIdx.x;
  const int wave = tid >> 5, lane = tid & 31;
  const int m0 = blockIdx.x * 128;
  const int n0 = blockIdx.y * 128;
  const int r = tid >> 1, off = (tid & 1) * 16;

  v8f acc[8];
#pragma unroll
  for (int i = 0; i < 8; ++i)
#pragma unroll
    for (int j = 0; j < 8; ++j) acc[i][j] = 0.0f;

  for (int k0 = 0; k0 < K; k0 += 32) {
    __syncthreads();
    const _Float16* pa = A + (size_t)(m0 + r) * lda + k0 + off;
    stage8(pa,     sA + r * 40 + off);
    stage8(pa + 8, sA + r * 40 + off + 8);
    const _Float16* pb = Bt + (size_t)(n0 + r) * ldb + k0 + off;
    stage8(pb,     sB + r * 40 + off);
    stage8(pb + 8, sB + r * 40 + off + 8);
    wait_async();
    __syncthreads();
    v16h a = load_a_frag(sA + wave * 16 * 40, 40, lane);
#pragma unroll
    for (int nt = 0; nt < 8; ++nt) {
      v16h b = load_b_frag(sB + nt * 16 * 40, 40, lane);
      acc[nt] = __builtin_amdgcn_wmma_f32_16x16x32_f16(false, a, false, b,
                                                       (short)0, acc[nt], false, false);
    }
  }

  const int mrow = m0 + wave * 16 + ((lane < 16) ? 0 : 8);
  const int nl = lane & 15;
#pragma unroll
  for (int nt = 0; nt < 8; ++nt) {
    int col = n0 + nt * 16 + nl;
    float bv = bias[col];
#pragma unroll
    for (int rr = 0; rr < 8; ++rr) {
      float x = swishf(acc[nt][rr] + bv);
      int row = mrow + rr;
      if (HASF) Cf[(size_t)row * ldcf + col] = x;
      if (HASH) Ch[(size_t)row * ldch + col] = (_Float16)x;
    }
  }
}

// ---------------------------------------------------------------------------
// Fused edge pipeline: per block = 128 edges = 16 dst nodes.
//   edge_in = [h[dst](128) | h[src](128) | tail(64, zero-padded)]  (K=320)
//   m = swish(edge_in @ W1 + b1); m2 = swish(m @ W2 + b2)
//   agg[node] = mean over the node's 8 contiguous edges -> upd_in[:,128:256]
// ---------------------------------------------------------------------------
__global__ __launch_bounds__(256)
void k_edge_msg(const _Float16* __restrict__ hh, int ldh,
                const _Float16* __restrict__ tail,
                const int* __restrict__ srcIdx,
                const _Float16* __restrict__ W1t, const float* __restrict__ b1,
                const _Float16* __restrict__ W2t, const float* __restrict__ b2,
                _Float16* __restrict__ agg, int ldagg, int aggOff) {
  __shared__ __align__(16) _Float16 sA[128 * 40];
  __shared__ __align__(16) _Float16 sB[128 * 40];
  __shared__ __align__(16) _Float16 sM[128 * 136];
  const int tid = threadIdx.x;
  const int wave = tid >> 5, lane = tid & 31;
  const int e0 = blockIdx.x * 128;
  const int r = tid >> 1, off = (tid & 1) * 16;
  const int e = e0 + r;
  const int se = srcIdx[e];
  const int de = e >> 3;  // dst = repeat(arange(N), DEG)

  v8f acc[8];
#pragma unroll
  for (int i = 0; i < 8; ++i)
#pragma unroll
    for (int j = 0; j < 8; ++j) acc[i][j] = 0.0f;

  // ---- GEMM 1: K = 320 ----
  for (int k0 = 0; k0 < 320; k0 += 32) {
    __syncthreads();
    const _Float16* pa;
    if (k0 < 128)       pa = hh + (size_t)de * ldh + k0 + off;
    else if (k0 < 256)  pa = hh + (size_t)se * ldh + (k0 - 128) + off;
    else                pa = tail + (size_t)e * 64 + (k0 - 256) + off;
    stage8(pa,     sA + r * 40 + off);
    stage8(pa + 8, sA + r * 40 + off + 8);
    const _Float16* pb = W1t + (size_t)r * 320 + k0 + off;
    stage8(pb,     sB + r * 40 + off);
    stage8(pb + 8, sB + r * 40 + off + 8);
    wait_async();
    __syncthreads();
    v16h a = load_a_frag(sA + wave * 16 * 40, 40, lane);
#pragma unroll
    for (int nt = 0; nt < 8; ++nt) {
      v16h b = load_b_frag(sB + nt * 16 * 40, 40, lane);
      acc[nt] = __builtin_amdgcn_wmma_f32_16x16x32_f16(false, a, false, b,
                                                       (short)0, acc[nt], false, false);
    }
  }

  // bias1 + swish -> sM (f16, row stride 136)
  {
    const int mrow = wave * 16 + ((lane < 16) ? 0 : 8);
    const int nl = lane & 15;
#pragma unroll
    for (int nt = 0; nt < 8; ++nt) {
      float bv = b1[nt * 16 + nl];
#pragma unroll
      for (int rr = 0; rr < 8; ++rr) {
        float x = swishf(acc[nt][rr] + bv);
        sM[(mrow + rr) * 136 + nt * 16 + nl] = (_Float16)x;
      }
    }
  }

  // ---- GEMM 2: K = 128, A from sM ----
  v8f acc2[8];
#pragma unroll
  for (int i = 0; i < 8; ++i)
#pragma unroll
    for (int j = 0; j < 8; ++j) acc2[i][j] = 0.0f;

  for (int k0 = 0; k0 < 128; k0 += 32) {
    __syncthreads();  // also guarantees sM writes visible on first iteration
    const _Float16* pb = W2t + (size_t)r * 128 + k0 + off;
    stage8(pb,     sB + r * 40 + off);
    stage8(pb + 8, sB + r * 40 + off + 8);
    wait_async();
    __syncthreads();
    v16h a = load_a_frag(sM + wave * 16 * 136 + k0, 136, lane);
#pragma unroll
    for (int nt = 0; nt < 8; ++nt) {
      v16h b = load_b_frag(sB + nt * 16 * 40, 40, lane);
      acc2[nt] = __builtin_amdgcn_wmma_f32_16x16x32_f16(false, a, false, b,
                                                        (short)0, acc2[nt], false, false);
    }
  }

  // bias2 + swish, then mean over each node's 8 edges.
  // Wave w rows 0..7 -> node 2w (lanes<16), rows 8..15 -> node 2w+1.
  {
    const int nl = lane & 15;
    const int node = (e0 >> 3) + wave * 2 + ((lane < 16) ? 0 : 1);
#pragma unroll
    for (int nt = 0; nt < 8; ++nt) {
      float bv = b2[nt * 16 + nl];
      float s = 0.0f;
#pragma unroll
      for (int rr = 0; rr < 8; ++rr) s += swishf(acc2[nt][rr] + bv);
      agg[(size_t)node * ldagg + aggOff + nt * 16 + nl] = (_Float16)(s * 0.125f);
    }
  }
}

// ---------------------------------------------------------------------------
// Prep kernels
// ---------------------------------------------------------------------------
__global__ void k_prep_w(const float* __restrict__ W, _Float16* __restrict__ Wt,
                         int fin, int fout, int finPad) {
  const int l = blockIdx.z;
  W += (size_t)l * fin * fout;
  Wt += (size_t)l * fout * finPad;
  int i = blockIdx.x * blockDim.x + threadIdx.x;
  if (i >= fout * finPad) return;
  int o = i / finPad, k = i % finPad;
  Wt[i] = (k < fin) ? (_Float16)W[(size_t)k * fout + o] : (_Float16)0.0f;
}

__global__ void k_prep_nodes(const float* __restrict__ u, const float* __restrict__ pos,
                             _Float16* __restrict__ node_in, _Float16* __restrict__ upd_in) {
  int n = blockIdx.x * blockDim.x + threadIdx.x;
  if (n >= N_NODES) return;
  _Float16* ni = node_in + (size_t)n * 64;
  for (int i = 0; i < 50; ++i) ni[i] = (_Float16)u[n * 50 + i];
  ni[50] = (_Float16)(pos[n * 2 + 1] * (1.0f / L_PDE));
  float var = pos[n * 2 + 0] * (1.0f / TMAX_C);
  ni[51] = (_Float16)var;
  for (int i = 52; i < 64; ++i) ni[i] = (_Float16)0.0f;
  _Float16* ui = upd_in + (size_t)n * 288;
  ui[256] = (_Float16)var;
  for (int i = 257; i < 288; ++i) ui[i] = (_Float16)0.0f;
}

__global__ void k_prep_tail(const float* __restrict__ u, const float* __restrict__ pos,
                            const int* __restrict__ ei, _Float16* __restrict__ tail) {
  int e = blockIdx.x * blockDim.x + threadIdx.x;
  if (e >= N_EDGES) return;
  int s = ei[e];
  int d = e >> 3;
  _Float16* t = tail + (size_t)e * 64;
  for (int i = 0; i < 50; ++i) t[i] = (_Float16)(u[d * 50 + i] - u[s * 50 + i]);
  t[50] = (_Float16)((pos[d * 2 + 1] - pos[s * 2 + 1]) * (1.0f / L_PDE));
  t[51] = (_Float16)(pos[d * 2 + 0] * (1.0f / TMAX_C));
  for (int i = 52; i < 64; ++i) t[i] = (_Float16)0.0f;
}

// ---------------------------------------------------------------------------
// GraphNorm: per-(graph,feature) mean/var over 4096 nodes.
// ---------------------------------------------------------------------------
__global__ void k_zero_stats(float* gs) {
  int i = blockIdx.x * blockDim.x + threadIdx.x;
  if (i < 2 * N_GRAPHS * HDIM) gs[i] = 0.0f;
}

__global__ __launch_bounds__(128)
void k_norm_a(const float* __restrict__ h, const float* __restrict__ up,
              float* __restrict__ hn, float* __restrict__ gstats) {
  const int f = threadIdx.x;
  const int n0 = blockIdx.x * 128;
  const int g = n0 >> 12;
  float s = 0.0f, ss = 0.0f;
  for (int i = 0; i < 128; ++i) {
    size_t idx = (size_t)(n0 + i) * HDIM + f;
    float v = h[idx] + up[idx];
    hn[idx] = v;
    s += v;
    ss += v * v;
  }
  atomicAdd(&gstats[g * HDIM + f], s);
  atomicAdd(&gstats[N_GRAPHS * HDIM + g * HDIM + f], ss);
}

__global__ void k_norm_b(const float* __restrict__ hn, const float* __restrict__ gstats,
                         float* __restrict__ h, _Float16* __restrict__ upd_in) {
  int i = blockIdx.x * blockDim.x + threadIdx.x;
  if (i >= N_NODES * HDIM) return;
  int n = i >> 7, f = i & 127;
  int g = n >> 12;
  float mu = gstats[g * HDIM + f] * (1.0f / NODES_PER_GRAPH);
  float var = gstats[N_GRAPHS * HDIM + g * HDIM + f] * (1.0f / NODES_PER_GRAPH) - mu * mu;
  float y = (hn[i] - mu) * rsqrtf(var + EPS_C);
  h[i] = y;
  upd_in[(size_t)n * 288 + f] = (_Float16)y;
}

// ---------------------------------------------------------------------------
// Decoder convs (tiny; scalar VALU)
// ---------------------------------------------------------------------------
__global__ void k_conv1(const float* __restrict__ x2, const float* __restrict__ W,
                        const float* __restrict__ b, float* __restrict__ y) {
  int i = blockIdx.x * blockDim.x + threadIdx.x;  // N * 8 * 38
  if (i >= N_NODES * 8 * 38) return;
  int p = i % 38;
  int oc = (i / 38) % 8;
  int n = i / (38 * 8);
  const float* x = x2 + (size_t)n * 256;
  float s = b[oc];
#pragma unroll
  for (int ci = 0; ci < 2; ++ci)
#pragma unroll
    for (int q = 0; q < 16; ++q)
      s += W[oc * 32 + ci * 16 + q] * x[ci * 128 + 3 * p + q];
  y[i] = swishf(s);
}

__global__ void k_conv2(const float* __restrict__ y, const float* __restrict__ W,
                        const float* __restrict__ b, const float* __restrict__ u,
                        float* __restrict__ out) {
  int i = blockIdx.x * blockDim.x + threadIdx.x;  // N * 2 * 25
  if (i >= N_NODES * 2 * TW) return;
  int t = i % TW;
  int c2 = (i / TW) & 1;
  int n = i / (2 * TW);
  const float* yr = y + (size_t)n * 304;
  float s = b[c2];
#pragma unroll
  for (int c1 = 0; c1 < 8; ++c1)
#pragma unroll
    for (int j = 0; j < 14; ++j)
      s += W[c2 * 112 + c1 * 14 + j] * yr[c1 * 38 + t + j];
  out[i] = u[i] + DT_C * (float)(t + 1) * s;
}

// ---------------------------------------------------------------------------
// Host launch
// ---------------------------------------------------------------------------
extern "C" void kernel_launch(void* const* d_in, const int* in_sizes, int n_in,
                              void* d_out, int out_size, void* d_ws, size_t ws_size,
                              hipStream_t stream) {
  (void)in_sizes; (void)n_in; (void)out_size; (void)ws_size;
  const float* u      = (const float*)d_in[0];
  const float* pos    = (const float*)d_in[1];
  const int*   ei     = (const int*)d_in[2];
  const float* embW1  = (const float*)d_in[4];
  const float* embB1  = (const float*)d_in[5];
  const float* embW2  = (const float*)d_in[6];
  const float* embB2  = (const float*)d_in[7];
  const float* msg1W  = (const float*)d_in[8];
  const float* msg1B  = (const float*)d_in[9];
  const float* msg2W  = (const float*)d_in[10];
  const float* msg2B  = (const float*)d_in[11];
  const float* upd1W  = (const float*)d_in[12];
  const float* upd1B  = (const float*)d_in[13];
  const float* upd2W  = (const float*)d_in[14];
  const float* upd2B  = (const float*)d_in[15];
  const float* dblW   = (const float*)d_in[16];
  const float* dblB   = (const float*)d_in[17];
  const float* c1W    = (const float*)d_in[18];
  const float* c1B    = (const float*)d_in[19];
  const float* c2W    = (const float*)d_in[20];
  const float* c2B    = (const float*)d_in[21];
  float* out = (float*)d_out;

  char* ws = (char*)d_ws;
  size_t o = 0;
  auto alloc = [&](size_t bytes) { size_t c = o; o = (o + bytes + 255) & ~(size_t)255; return c; };
  size_t o_embW1t = alloc(128 * 64 * 2);
  size_t o_embW2t = alloc(128 * 128 * 2);
  size_t o_msg1t  = alloc((size_t)N_LAYERS * 128 * 320 * 2);
  size_t o_msg2t  = alloc((size_t)N_LAYERS * 128 * 128 * 2);
  size_t o_upd1t  = alloc((size_t)N_LAYERS * 128 * 288 * 2);
  size_t o_upd2t  = alloc((size_t)N_LAYERS * 128 * 128 * 2);
  size_t o_dblt   = alloc(256 * 128 * 2);
  size_t o_gstats = alloc(2 * N_GRAPHS * HDIM * 4);
  size_t o_nodein = alloc((size_t)N_NODES * 64 * 2);
  size_t o_tail   = alloc((size_t)N_EDGES * 64 * 2);
  size_t o_updin  = alloc((size_t)N_NODES * 288 * 2);
  size_t o_tmpn   = alloc((size_t)N_NODES * 128 * 2);
  size_t o_h      = alloc((size_t)N_NODES * 128 * 4);
  size_t o_up     = alloc((size_t)N_NODES * 128 * 4);
  size_t o_hn     = alloc((size_t)N_NODES * 128 * 4);
  size_t o_x2     = o_tail;  // alias: tail dead after last edge kernel (N*256*4 == E*64*2)
  size_t o_ybuf   = o_h;     // alias: h/up/hn dead in decoder (N*304*4 < 3*N*128*4)

  _Float16* embW1t = (_Float16*)(ws + o_embW1t);
  _Float16* embW2t = (_Float16*)(ws + o_embW2t);
  _Float16* msg1t  = (_Float16*)(ws + o_msg1t);
  _Float16* msg2t  = (_Float16*)(ws + o_msg2t);
  _Float16* upd1t  = (_Float16*)(ws + o_upd1t);
  _Float16* upd2t  = (_Float16*)(ws + o_upd2t);
  _Float16* dblt   = (_Float16*)(ws + o_dblt);
  float*    gstats = (float*)(ws + o_gstats);
  _Float16* nodein = (_Float16*)(ws + o_nodein);
  _Float16* tail   = (_Float16*)(ws + o_tail);
  _Float16* updin  = (_Float16*)(ws + o_updin);
  _Float16* tmpn   = (_Float16*)(ws + o_tmpn);
  float*    hbuf   = (float*)(ws + o_h);
  float*    upbuf  = (float*)(ws + o_up);
  float*    hnbuf  = (float*)(ws + o_hn);
  float*    x2buf  = (float*)(ws + o_x2);
  float*    ybuf   = (float*)(ws + o_ybuf);

  // --- weight prep (transpose + f16 + K-pad) ---
  auto prepW = [&](const float* W, _Float16* Wt, int fin, int fout, int fpad, int nl) {
    dim3 g((fout * fpad + 255) / 256, 1, nl);
    k_prep_w<<<g, 256, 0, stream>>>(W, Wt, fin, fout, fpad);
  };
  prepW(embW1, embW1t, 52, 128, 64, 1);
  prepW(embW2, embW2t, 128, 128, 128, 1);
  prepW(msg1W, msg1t, 308, 128, 320, N_LAYERS);
  prepW(msg2W, msg2t, 128, 128, 128, N_LAYERS);
  prepW(upd1W, upd1t, 257, 128, 288, N_LAYERS);
  prepW(upd2W, upd2t, 128, 128, 128, N_LAYERS);
  prepW(dblW, dblt, 128, 256, 128, 1);

  k_prep_nodes<<<(N_NODES + 255) / 256, 256, 0, stream>>>(u, pos, nodein, updin);
  k_prep_tail<<<(N_EDGES + 255) / 256, 256, 0, stream>>>(u, pos, ei, tail);

  const dim3 gN(N_NODES / 128, 1);
  // --- embedding: h = swish(swish(node_in@W1+b1)@W2+b2); f16 copy -> upd_in[:,0:128]
  k_gemm<false, true><<<gN, 256, 0, stream>>>(nodein, 64, embW1t, 64, embB1,
                                              (float*)nullptr, 0, tmpn, 128, 64);
  k_gemm<true, true><<<gN, 256, 0, stream>>>(tmpn, 128, embW2t, 128, embB2,
                                             hbuf, 128, updin, 288, 128);

  // --- message-passing layers ---
  for (int l = 0; l < N_LAYERS; ++l) {
    k_edge_msg<<<dim3(N_EDGES / 128), 256, 0, stream>>>(
        updin, 288, tail, ei,
        msg1t + (size_t)l * 128 * 320, msg1B + l * 128,
        msg2t + (size_t)l * 128 * 128, msg2B + l * 128,
        updin, 288, 128);
    k_gemm<false, true><<<gN, 256, 0, stream>>>(updin, 288, upd1t + (size_t)l * 128 * 288, 288,
                                                upd1B + l * 128, (float*)nullptr, 0, tmpn, 128, 288);
    k_gemm<true, false><<<gN, 256, 0, stream>>>(tmpn, 128, upd2t + (size_t)l * 128 * 128, 128,
                                                upd2B + l * 128, upbuf, 128, (_Float16*)nullptr, 0, 128);
    k_zero_stats<<<(2 * N_GRAPHS * HDIM + 255) / 256, 256, 0, stream>>>(gstats);
    k_norm_a<<<N_NODES / 128, 128, 0, stream>>>(hbuf, upbuf, hnbuf, gstats);
    k_norm_b<<<(N_NODES * HDIM + 255) / 256, 256, 0, stream>>>(hnbuf, gstats, hbuf, updin);
  }

  // --- decoder ---
  k_gemm<true, false><<<dim3(N_NODES / 128, 2), 256, 0, stream>>>(updin, 288, dblt, 128, dblB,
                                                                  x2buf, 256, (_Float16*)nullptr, 0, 128);
  k_conv1<<<(N_NODES * 8 * 38 + 255) / 256, 256, 0, stream>>>(x2buf, c1W, c1B, ybuf);
  k_conv2<<<(N_NODES * 2 * TW + 255) / 256, 256, 0, stream>>>(ybuf, c2W, c2B, u, out);
}
